// CausalSelfAttention_89069031785072
// MI455X (gfx1250) — matrix operand order repaired and checked by reading, not compiled, not run on hardware
//
#include <hip/hip_runtime.h>
#include <hip/hip_bf16.h>

typedef __bf16 bf16_t;
typedef __attribute__((ext_vector_type(16))) __bf16 v16bf;
typedef __attribute__((ext_vector_type(8)))  __bf16 v8bf;
typedef __attribute__((ext_vector_type(8)))  float  v8f;

#define BB   4
#define TT   2048
#define CC   1024
#define HH   16
#define DD   64
#define N3   3072
#define BT   (BB*TT)      // 8192 rows

// ---------------- WMMA operand loaders (layouts per CDNA5 ISA) ----

// A-matrix 16x32 bf16: lane row = (row base + lane%16); elements i=0..15 -> K =
// k0 + half*16 + i (contiguous). Two 16B loads.
__device__ __forceinline__ v16bf load_A16x32(const bf16_t* A, int lda, int row,
                                             int k0, int half) {
  const bf16_t* p = A + (size_t)row * lda + k0 + half * 16;
  v8bf c0 = *(const v8bf*)p;
  v8bf c1 = *(const v8bf*)(p + 8);
  v16bf r;
#pragma unroll
  for (int i = 0; i < 8; i++) { r[i] = c0[i]; r[i + 8] = c1[i]; }
  return r;
}

// B-matrix 32x16 bf16 from K-contiguous ("transposed") storage Bt[n*ldb + k]:
// lane col = col0 + lane%16 ; elements i -> K = k0 + half*16 + i (contiguous).
__device__ __forceinline__ v16bf load_B32x16(const bf16_t* Bt, int ldb, int col0,
                                             int k0, int l16, int half) {
  const bf16_t* p = Bt + (size_t)(col0 + l16) * ldb + k0 + half * 16;
  return *(const v16bf*)p;
}

__device__ __forceinline__ v8f wmma_bf16(v16bf a, v16bf b, v8f c) {
  return __builtin_amdgcn_wmma_f32_16x16x32_bf16(false, a, false, b, (short)0,
                                                 c, false, false);
}

__device__ __forceinline__ float rmax16(float v) {
#pragma unroll
  for (int m = 1; m < 16; m <<= 1) v = fmaxf(v, __shfl_xor(v, m, 32));
  return v;
}
__device__ __forceinline__ float rsum16(float v) {
#pragma unroll
  for (int m = 1; m < 16; m <<= 1) v += __shfl_xor(v, m, 32);
  return v;
}

// ---------------- conversion kernels ----------------------------------------

__global__ void cvt_f32_bf16(const float* __restrict__ in, bf16_t* __restrict__ out,
                             int n) {
  int i = blockIdx.x * blockDim.x + threadIdx.x;
  if (i < n) out[i] = (bf16_t)in[i];
}

// W[K,N] fp32 row-major -> Wt[N,K] bf16 (makes WMMA B loads K-contiguous)
__global__ void transpose_f32_bf16(const float* __restrict__ W, bf16_t* __restrict__ Wt,
                                   int K, int N) {
  int i = blockIdx.x * blockDim.x + threadIdx.x;
  if (i < K * N) {
    int n = i / K, k = i - n * K;
    Wt[(size_t)n * K + k] = (bf16_t)W[(size_t)k * N + n];
  }
}

// ---------------- QKV projection GEMM + head scatter -------------------------
// [BT,CC] x [CC,N3] ; per-wave tile 32(M) x 64(N): 8 accs, B reused 2x, A 4x.
__global__ void __launch_bounds__(256) qkv_gemm(const bf16_t* __restrict__ xh,
                                                const bf16_t* __restrict__ WaT,
                                                bf16_t* __restrict__ qh,
                                                bf16_t* __restrict__ kh,
                                                bf16_t* __restrict__ vTh) {
  const int lane = threadIdx.x & 31;
  const int half = lane >> 4;
  const int l16  = lane & 15;
  const int wid  = threadIdx.x >> 5;
  const int ntiles = N3 / 64;                       // 48
  int gw = blockIdx.x * 8 + wid;
  int mt = gw / ntiles, nt = gw - mt * ntiles;
  if (mt >= BT / 32) return;
  int m0 = mt * 32, n0 = nt * 64;

  v8f zero = {0.f, 0.f, 0.f, 0.f, 0.f, 0.f, 0.f, 0.f};
  v8f acc[2][4] = {{zero, zero, zero, zero}, {zero, zero, zero, zero}};

  for (int k0 = 0; k0 < CC; k0 += 32) {
    if (k0 + 32 < CC) {   // pull next K-step lines L2 -> WGP$ (global_prefetch_b8)
      __builtin_prefetch(xh + (size_t)(m0 + l16) * CC + k0 + 32, 0, 3);
      __builtin_prefetch(WaT + (size_t)(n0 + l16) * CC + k0 + 32, 0, 3);
    }
    v16bf a0 = load_A16x32(xh, CC, m0 + l16,      k0, half);
    v16bf a1 = load_A16x32(xh, CC, m0 + 16 + l16, k0, half);
#pragma unroll
    for (int j = 0; j < 4; j++) {
      v16bf b = load_B32x16(WaT, CC, n0 + j * 16, k0, l16, half);
      acc[0][j] = wmma_bf16(a0, b, acc[0][j]);
      acc[1][j] = wmma_bf16(a1, b, acc[1][j]);
    }
  }

  int section = n0 / CC;                             // 0=q 1=k 2=v (64 | 1024)
#pragma unroll
  for (int ms = 0; ms < 2; ms++) {
#pragma unroll
    for (int j = 0; j < 4; j++) {
      int n = n0 + j * 16 + l16;
      int f = n % CC;
      int h = f / DD, d = f - h * DD;
#pragma unroll
      for (int r = 0; r < 8; r++) {
        int trow = m0 + ms * 16 + r + half * 8;      // global token row
        int b    = trow / TT, t = trow - b * TT;
        size_t hb = (size_t)(b * HH + h);
        float v = acc[ms][j][r];
        if (section == 0)      qh[(hb * TT + t) * DD + d]  = (bf16_t)(v * 0.03125f);
        else if (section == 1) kh[(hb * TT + t) * DD + d]  = (bf16_t)v;
        else                   vTh[(hb * DD + d) * TT + t] = (bf16_t)v;
      }
    }
  }
}

// ---------------- flash attention: one wave per (b,h, 32-row q block) --------
// Per 32-key block: 4 K-frag loads -> 8 QK WMMAs; 4 V-frag loads -> 8 PV WMMAs.
__global__ void __launch_bounds__(256) attn_kernel(const bf16_t* __restrict__ qh,
                                                   const bf16_t* __restrict__ kh,
                                                   const bf16_t* __restrict__ vTh,
                                                   bf16_t* __restrict__ yh) {
  __shared__ bf16_t plds[8][32 * 32];                // per-wave P tile (bf16)
  const int lane = threadIdx.x & 31;
  const int half = lane >> 4;
  const int l16  = lane & 15;
  const int wid  = threadIdx.x >> 5;
  int gw = blockIdx.x * 8 + wid;
  int qt2 = gw % (TT / 32);                          // 32-row query block
  int bh  = gw / (TT / 32);
  if (bh >= BB * HH) return;
  int q0 = qt2 * 32;

  const bf16_t* qb = qh + (size_t)bh * TT * DD;
  const bf16_t* kb = kh + (size_t)bh * TT * DD;
  const bf16_t* vb = vTh + (size_t)bh * DD * TT;
  bf16_t* myp = plds[wid];

  // Q block 32x64 in A-layout registers: [msub][kstep]
  v16bf aQ[2][2];
#pragma unroll
  for (int ms = 0; ms < 2; ms++) {
    aQ[ms][0] = load_A16x32(qb, DD, q0 + ms * 16 + l16, 0, half);
    aQ[ms][1] = load_A16x32(qb, DD, q0 + ms * 16 + l16, 32, half);
  }

  v8f zero = {0.f, 0.f, 0.f, 0.f, 0.f, 0.f, 0.f, 0.f};
  v8f O[2][4] = {{zero, zero, zero, zero}, {zero, zero, zero, zero}};
  float mrun[2][8], lrun[2][8];
#pragma unroll
  for (int ms = 0; ms < 2; ms++)
#pragma unroll
    for (int r = 0; r < 8; r++) { mrun[ms][r] = -__builtin_inff(); lrun[ms][r] = 0.f; }

  const int nblk = qt2 + 1;                          // causal 32-key blocks
  for (int blk = 0; blk < nblk; ++blk) {
    int s0 = blk * 32;
    if (blk + 1 < nblk) {  // prefetch next key block K/V lines into WGP$
      __builtin_prefetch(kb + (size_t)(s0 + 32 + l16) * DD, 0, 3);
      __builtin_prefetch(vb + (size_t)(l16) * TT + s0 + 32, 0, 3);
    }
    // S = Q @ K^T : K fragments loaded once, shared across both M sub-tiles
    v8f S[2][2] = {{zero, zero}, {zero, zero}};      // [msub][key half]
#pragma unroll
    for (int ks = 0; ks < 2; ks++) {                 // K-dim step (d 0..31/32..63)
#pragma unroll
      for (int ns = 0; ns < 2; ns++) {               // key half
        v16bf bK = load_B32x16(kb, DD, s0 + ns * 16, ks * 32, l16, half);
        S[0][ns] = wmma_bf16(aQ[0][ks], bK, S[0][ns]);
        S[1][ns] = wmma_bf16(aQ[1][ks], bK, S[1][ns]);
      }
    }

#pragma unroll
    for (int ms = 0; ms < 2; ms++) {
#pragma unroll
      for (int r = 0; r < 8; r++) {
        int trow = q0 + ms * 16 + r + half * 8;
        float s0v = (s0 + l16      <= trow) ? S[ms][0][r] : -__builtin_inff();
        float s1v = (s0 + 16 + l16 <= trow) ? S[ms][1][r] : -__builtin_inff();
        float mx   = rmax16(fmaxf(s0v, s1v));
        float newm = fmaxf(mrun[ms][r], mx);
        float sc   = __expf(mrun[ms][r] - newm);     // 0 on first block
        float p0   = __expf(s0v - newm);
        float p1   = __expf(s1v - newm);
        lrun[ms][r] = lrun[ms][r] * sc + rsum16(p0 + p1);
        mrun[ms][r] = newm;
#pragma unroll
        for (int j = 0; j < 4; j++) O[ms][j][r] *= sc;
        int m = ms * 16 + r + half * 8;              // C-layout row in LDS tile
        myp[m * 32 + l16]      = (bf16_t)p0;
        myp[m * 32 + 16 + l16] = (bf16_t)p1;
      }
    }
    asm volatile("s_wait_dscnt 0" ::: "memory");     // LDS RAW within wave

    // O += P(32x32) @ V(32x64) ; V fragments shared across both M sub-tiles
    v16bf aP0 = load_A16x32(myp, 32, l16,      0, half);
    v16bf aP1 = load_A16x32(myp, 32, 16 + l16, 0, half);
#pragma unroll
    for (int j = 0; j < 4; j++) {
      v16bf bV = load_B32x16(vb, TT, j * 16, s0, l16, half);
      O[0][j] = wmma_bf16(aP0, bV, O[0][j]);
      O[1][j] = wmma_bf16(aP1, bV, O[1][j]);
    }
  }

  // normalize and write y as [BT, CC] bf16 (A-operand for proj GEMM)
  int b = bh / HH, h = bh - b * HH;
#pragma unroll
  for (int ms = 0; ms < 2; ms++)
#pragma unroll
    for (int j = 0; j < 4; j++)
#pragma unroll
      for (int r = 0; r < 8; r++) {
        int row = q0 + ms * 16 + r + half * 8;
        float v = O[ms][j][r] / lrun[ms][r];
        yh[((size_t)b * TT + row) * CC + h * DD + j * 16 + l16] = (bf16_t)v;
      }
}

// ---------------- output projection GEMM -> fp32 -----------------------------
__global__ void __launch_bounds__(256) proj_gemm(const bf16_t* __restrict__ yh,
                                                 const bf16_t* __restrict__ WpT,
                                                 float* __restrict__ out) {
  const int lane = threadIdx.x & 31;
  const int half = lane >> 4;
  const int l16  = lane & 15;
  const int wid  = threadIdx.x >> 5;
  const int ntiles = CC / 64;                        // 16
  int gw = blockIdx.x * 8 + wid;
  int mt = gw / ntiles, nt = gw - mt * ntiles;
  if (mt >= BT / 32) return;
  int m0 = mt * 32, n0 = nt * 64;

  v8f zero = {0.f, 0.f, 0.f, 0.f, 0.f, 0.f, 0.f, 0.f};
  v8f acc[2][4] = {{zero, zero, zero, zero}, {zero, zero, zero, zero}};

  for (int k0 = 0; k0 < CC; k0 += 32) {
    if (k0 + 32 < CC) {
      __builtin_prefetch(yh + (size_t)(m0 + l16) * CC + k0 + 32, 0, 3);
      __builtin_prefetch(WpT + (size_t)(n0 + l16) * CC + k0 + 32, 0, 3);
    }
    v16bf a0 = load_A16x32(yh, CC, m0 + l16,      k0, half);
    v16bf a1 = load_A16x32(yh, CC, m0 + 16 + l16, k0, half);
#pragma unroll
    for (int j = 0; j < 4; j++) {
      v16bf b = load_B32x16(WpT, CC, n0 + j * 16, k0, l16, half);
      acc[0][j] = wmma_bf16(a0, b, acc[0][j]);
      acc[1][j] = wmma_bf16(a1, b, acc[1][j]);
    }
  }
#pragma unroll
  for (int ms = 0; ms < 2; ms++)
#pragma unroll
    for (int j = 0; j < 4; j++)
#pragma unroll
      for (int r = 0; r < 8; r++)
        out[(size_t)(m0 + ms * 16 + r + half * 8) * CC + n0 + j * 16 + l16] =
            acc[ms][j][r];
}

// ---------------- launcher ---------------------------------------------------
extern "C" void kernel_launch(void* const* d_in, const int* in_sizes, int n_in,
                              void* d_out, int out_size, void* d_ws, size_t ws_size,
                              hipStream_t stream) {
  (void)in_sizes; (void)n_in; (void)out_size; (void)ws_size;
  const float* x  = (const float*)d_in[0];
  const float* Wa = (const float*)d_in[1];
  const float* Wp = (const float*)d_in[2];
  float* out = (float*)d_out;

  char* ws = (char*)d_ws;
  size_t o = 0;
  bf16_t* xh  = (bf16_t*)(ws + o); o += (size_t)BT * CC * 2;   // 16 MiB
  bf16_t* WaT = (bf16_t*)(ws + o); o += (size_t)N3 * CC * 2;   //  6 MiB
  bf16_t* WpT = (bf16_t*)(ws + o); o += (size_t)CC * CC * 2;   //  2 MiB
  bf16_t* qh  = (bf16_t*)(ws + o); o += (size_t)BT * CC * 2;   // 16 MiB
  bf16_t* kh  = (bf16_t*)(ws + o); o += (size_t)BT * CC * 2;   // 16 MiB
  bf16_t* vTh = (bf16_t*)(ws + o); o += (size_t)BT * CC * 2;   // 16 MiB
  bf16_t* yh  = (bf16_t*)(ws + o); o += (size_t)BT * CC * 2;   // 16 MiB

  int ncvt = BT * CC;
  cvt_f32_bf16<<<(ncvt + 255) / 256, 256, 0, stream>>>(x, xh, ncvt);
  transpose_f32_bf16<<<(CC * N3 + 255) / 256, 256, 0, stream>>>(Wa, WaT, CC, N3);
  transpose_f32_bf16<<<(CC * CC + 255) / 256, 256, 0, stream>>>(Wp, WpT, CC, CC);

  // QKV: (8192/32)*(3072/64)=12288 wave-tiles, 8 waves/block
  qkv_gemm<<<12288 / 8, 256, 0, stream>>>(xh, WaT, qh, kh, vTh);
  // attention: B*H*(T/32)=4096 waves
  attn_kernel<<<4096 / 8, 256, 0, stream>>>(qh, kh, vTh, yh);
  // proj: (8192/32)*(1024/64)=4096 wave-tiles
  proj_gemm<<<4096 / 8, 256, 0, stream>>>(yh, WpT, out);
}
